// PrivacyLoss_3770981285903
// MI455X (gfx1250) — matrix operand order repaired and checked by reading, not compiled, not run on hardware
//
#include <hip/hip_runtime.h>
#include <hip/hip_bf16.h>

// ---------------------------------------------------------------------------
// CDNA5 (gfx1250) bf16x3 WMMA implementation of the retrieval/privacy loss.
//   emb = X @ W        (2048x12288 @ 12288x512)  via v_wmma_f32_16x16x32_bf16
//   sq  = |e|^2 + |l|^2 - 2 e.l  (2048 x 100000) via v_wmma, fused min-reduce
//   out[b] = mse + min(50 - sqrt(min_sq[b]), 0) * 5
// fp32 tiles are split ONCE during LDS staging into (hi, lo) bf16 tiles; the
// inner loop is pure ds_load_b128 fragment loads + WMMA (bf16x3: hh+hl+lh).
// ---------------------------------------------------------------------------

typedef __attribute__((ext_vector_type(16))) __bf16 v16bf;
typedef __attribute__((ext_vector_type(8)))  __bf16 v8bf;
typedef __attribute__((ext_vector_type(8)))  float  v8f;

#define INF_BITS 0x7F800000u

#define BM 128
#define BN 128
#define BK 32
#define LDB 40   // bf16 LDS tile stride: 80 bytes/row -> 16B-aligned rows,
                 // 20-dword bank step => 16 consecutive rows hit 16 banks.

__device__ __forceinline__ __bf16 bits2bf(unsigned short s) {
  union { unsigned short u; __bf16 b; } cv; cv.u = s; return cv.b;
}

// Pack hi-bf16 (truncation) of 4 floats into 4 packed bf16; residuals out.
__device__ __forceinline__ uint2 pack_hi4(float4 v, float4 &lo) {
  unsigned ux = __float_as_uint(v.x) & 0xFFFF0000u;
  unsigned uy = __float_as_uint(v.y) & 0xFFFF0000u;
  unsigned uz = __float_as_uint(v.z) & 0xFFFF0000u;
  unsigned uw = __float_as_uint(v.w) & 0xFFFF0000u;
  lo.x = v.x - __uint_as_float(ux);
  lo.y = v.y - __uint_as_float(uy);
  lo.z = v.z - __uint_as_float(uz);
  lo.w = v.w - __uint_as_float(uw);
  return make_uint2((ux >> 16) | uy, (uz >> 16) | uw);
}
__device__ __forceinline__ uint2 pack_bf4(float4 v) {
  return make_uint2((__float_as_uint(v.x) >> 16) | (__float_as_uint(v.y) & 0xFFFF0000u),
                    (__float_as_uint(v.z) >> 16) | (__float_as_uint(v.w) & 0xFFFF0000u));
}

// A fragment 16x32: lane = row, K {base..base+7, base+16..base+23}, base=8*(l>=16).
__device__ __forceinline__ v16bf frag_a16(const __bf16* s, int rBase, int lane) {
  const __bf16* p = s + (rBase + (lane & 15)) * LDB + ((lane >> 4) << 3);
  v8bf a = *(const v8bf*)p;
  v8bf b = *(const v8bf*)(p + 16);
  return __builtin_shufflevector(a, b, 0,1,2,3,4,5,6,7,8,9,10,11,12,13,14,15);
}
// B fragment 32x16 from [n][k] tile: lane = column, K {base..base+7, base+16..base+23}, base=8*(l>=16).
__device__ __forceinline__ v16bf frag_b16(const __bf16* s, int rBase, int lane) {
  const __bf16* p = s + (rBase + (lane & 15)) * LDB + ((lane >> 4) << 3);
  v8bf a = *(const v8bf*)p;
  v8bf b = *(const v8bf*)(p + 16);
  return __builtin_shufflevector(a, b, 0,1,2,3,4,5,6,7,8,9,10,11,12,13,14,15);
}

// D = A*B + C with bf16x3 error compensation: hi*hi + hi*lo + lo*hi.
__device__ __forceinline__ v8f wmma3(v16bf ah, v16bf al, v16bf bh, v16bf bl, v8f c) {
  c = __builtin_amdgcn_wmma_f32_16x16x32_bf16(false, ah, false, bh, (short)0, c, false, false);
  c = __builtin_amdgcn_wmma_f32_16x16x32_bf16(false, ah, false, bl, (short)0, c, false, false);
  c = __builtin_amdgcn_wmma_f32_16x16x32_bf16(false, al, false, bh, (short)0, c, false, false);
  return c;
}

// ---------------------------------------------------------------------------
__global__ void k_init(unsigned* __restrict__ minBits, float* __restrict__ mseAcc, int B) {
  int i = blockIdx.x * blockDim.x + threadIdx.x;
  if (i == 0) *mseAcc = 0.0f;
  if (i < B) minBits[i] = INF_BITS;
}

__global__ void k_mse(const float4* __restrict__ x, const float4* __restrict__ y,
                      float* __restrict__ acc, long long n4) {
  long long i      = (long long)blockIdx.x * blockDim.x + threadIdx.x;
  long long stride = (long long)gridDim.x * blockDim.x;
  float s = 0.0f;
  for (; i < n4; i += stride) {
    float4 a = x[i], b = y[i];
    float dx = a.x - b.x, dy = a.y - b.y, dz = a.z - b.z, dw = a.w - b.w;
    s += dx * dx + dy * dy + dz * dz + dw * dw;
  }
#pragma unroll
  for (int off = 16; off > 0; off >>= 1) s += __shfl_xor(s, off, 32);
  __shared__ float ps[8];
  if ((threadIdx.x & 31) == 0) ps[threadIdx.x >> 5] = s;
  __syncthreads();
  if (threadIdx.x == 0) {
    float t = 0.0f;
    for (int w = 0; w < 8; ++w) t += ps[w];
    atomicAdd(acc, t);
  }
}

// One block (128 threads) per row of length 512: out[row] = sum(row^2).
__global__ void k_rownorm(const float* __restrict__ A, float* __restrict__ out, int cols) {
  int row = blockIdx.x;
  const float4* p = (const float4*)(A + (size_t)row * cols);
  float4 v = p[threadIdx.x];
  float s = v.x * v.x + v.y * v.y + v.z * v.z + v.w * v.w;
#pragma unroll
  for (int off = 16; off > 0; off >>= 1) s += __shfl_xor(s, off, 32);
  __shared__ float ps[4];
  if ((threadIdx.x & 31) == 0) ps[threadIdx.x >> 5] = s;
  __syncthreads();
  if (threadIdx.x == 0) out[row] = ps[0] + ps[1] + ps[2] + ps[3];
}

// GEMM1: emb[M,N] = X[M,K] @ W[K,N].  M=2048, N=512, K=12288.
__global__ __launch_bounds__(256) void k_gemm1(const float* __restrict__ X,
                                               const float* __restrict__ W,
                                               float* __restrict__ emb,
                                               int M, int N, int K) {
  __shared__ __bf16 sAh[BM * LDB], sAl[BM * LDB];  // X tile  [m][k]
  __shared__ __bf16 sBh[BN * LDB], sBl[BN * LDB];  // W tile transposed [n][k]
  const int m0   = blockIdx.x * BM;
  const int n0   = blockIdx.y * BN;
  const int tid  = threadIdx.x;
  const int lane = tid & 31;
  const int wave = tid >> 5;       // 8 waves: 2 (M) x 4 (N), 64x32 each
  const int mw   = (wave & 1) * 64;
  const int nw   = (wave >> 1) * 32;

  v8f acc[4][2];
#pragma unroll
  for (int i = 0; i < 4; ++i)
#pragma unroll
    for (int j = 0; j < 2; ++j)
#pragma unroll
      for (int r = 0; r < 8; ++r) acc[i][j][r] = 0.0f;

  float4 ra[4], rb[4];
#pragma unroll
  for (int i = 0; i < 4; ++i) {   // prefetch tile 0
    int idx = tid + (i << 8);
    ra[i] = *(const float4*)(X + (size_t)(m0 + (idx >> 3)) * K + ((idx & 7) << 2));
    rb[i] = *(const float4*)(W + (size_t)(idx >> 5) * N + n0 + ((idx & 31) << 2));
  }

  for (int k0 = 0; k0 < K; k0 += BK) {
    // stage tiles: convert fp32 -> (hi, lo) bf16 once per element
#pragma unroll
    for (int i = 0; i < 4; ++i) {
      int idx = tid + (i << 8);
      int row = idx >> 3, q = (idx & 7) << 2;
      float4 lo; uint2 h = pack_hi4(ra[i], lo); uint2 l = pack_bf4(lo);
      *(uint2*)&sAh[row * LDB + q] = h;
      *(uint2*)&sAl[row * LDB + q] = l;
    }
#pragma unroll
    for (int i = 0; i < 4; ++i) {  // transpose W tile into [n][k]
      int idx = tid + (i << 8);
      int kr = idx >> 5, nq = (idx & 31) << 2;
      float c[4] = {rb[i].x, rb[i].y, rb[i].z, rb[i].w};
#pragma unroll
      for (int cc = 0; cc < 4; ++cc) {
        unsigned u  = __float_as_uint(c[cc]);
        unsigned hu = u & 0xFFFF0000u;
        float    lf = c[cc] - __uint_as_float(hu);
        sBh[(nq + cc) * LDB + kr] = bits2bf((unsigned short)(hu >> 16));
        sBl[(nq + cc) * LDB + kr] = bits2bf((unsigned short)(__float_as_uint(lf) >> 16));
      }
    }
    __syncthreads();

    if (k0 + BK < K) {             // prefetch next tile into registers
#pragma unroll
      for (int i = 0; i < 4; ++i) {
        int idx = tid + (i << 8);
        ra[i] = *(const float4*)(X + (size_t)(m0 + (idx >> 3)) * K + k0 + BK + ((idx & 7) << 2));
        rb[i] = *(const float4*)(W + (size_t)(k0 + BK + (idx >> 5)) * N + n0 + ((idx & 31) << 2));
      }
    }

    v16bf ah[4], al[4], bh[2], bl[2];
#pragma unroll
    for (int i = 0; i < 4; ++i) {
      ah[i] = frag_a16(sAh, mw + i * 16, lane);
      al[i] = frag_a16(sAl, mw + i * 16, lane);
    }
#pragma unroll
    for (int j = 0; j < 2; ++j) {
      bh[j] = frag_b16(sBh, nw + j * 16, lane);
      bl[j] = frag_b16(sBl, nw + j * 16, lane);
    }
#pragma unroll
    for (int i = 0; i < 4; ++i)
#pragma unroll
      for (int j = 0; j < 2; ++j) acc[i][j] = wmma3(ah[i], al[i], bh[j], bl[j], acc[i][j]);
    __syncthreads();
  }

#pragma unroll
  for (int i = 0; i < 4; ++i)
#pragma unroll
    for (int j = 0; j < 2; ++j) {
      int col   = n0 + nw + j * 16 + (lane & 15);
      int rbase = m0 + mw + i * 16 + ((lane >> 4) << 3);
#pragma unroll
      for (int r = 0; r < 8; ++r)
        emb[(size_t)(rbase + r) * N + col] = acc[i][j][r];
    }
}

// GEMM2 fused: sq = |e|^2 + |l|^2 - 2 e.l, min over the N tile, atomicMin.
__global__ __launch_bounds__(256) void k_dist(const float* __restrict__ emb,
                                              const float* __restrict__ lut,
                                              const float* __restrict__ enorm,
                                              const float* __restrict__ lnorm,
                                              unsigned* __restrict__ minBits,
                                              int M, int N, int K) {
  __shared__ __bf16 sAh[BM * LDB], sAl[BM * LDB];  // emb tile [m][k]
  __shared__ __bf16 sBh[BN * LDB], sBl[BN * LDB];  // lookup tile [n][k]
  __shared__ unsigned smin[BM];
  const int m0   = blockIdx.x * BM;  // x fastest: 16 M-tiles share a lookup tile in L2
  const int n0   = blockIdx.y * BN;
  const int tid  = threadIdx.x;
  const int lane = tid & 31;
  const int wave = tid >> 5;
  const int mw   = (wave & 1) * 64;
  const int nw   = (wave >> 1) * 32;

  if (tid < BM) smin[tid] = INF_BITS;

  v8f acc[4][2];
#pragma unroll
  for (int i = 0; i < 4; ++i)
#pragma unroll
    for (int j = 0; j < 2; ++j)
#pragma unroll
      for (int r = 0; r < 8; ++r) acc[i][j][r] = 0.0f;

  float4 ra[4], rb[4];
#pragma unroll
  for (int i = 0; i < 4; ++i) {   // prefetch tile 0
    int idx = tid + (i << 8);
    int row = idx >> 3, q = (idx & 7) << 2;
    int gr  = n0 + row; gr = (gr < N) ? gr : (N - 1);
    ra[i] = *(const float4*)(emb + (size_t)(m0 + row) * K + q);
    rb[i] = *(const float4*)(lut + (size_t)gr * K + q);
  }

  for (int k0 = 0; k0 < K; k0 += BK) {
#pragma unroll
    for (int i = 0; i < 4; ++i) {
      int idx = tid + (i << 8);
      int row = idx >> 3, q = (idx & 7) << 2;
      float4 lo; uint2 h, l;
      h = pack_hi4(ra[i], lo); l = pack_bf4(lo);
      *(uint2*)&sAh[row * LDB + q] = h;
      *(uint2*)&sAl[row * LDB + q] = l;
      h = pack_hi4(rb[i], lo); l = pack_bf4(lo);
      *(uint2*)&sBh[row * LDB + q] = h;
      *(uint2*)&sBl[row * LDB + q] = l;
    }
    __syncthreads();

    if (k0 + BK < K) {
#pragma unroll
      for (int i = 0; i < 4; ++i) {
        int idx = tid + (i << 8);
        int row = idx >> 3, q = (idx & 7) << 2;
        int gr  = n0 + row; gr = (gr < N) ? gr : (N - 1);
        ra[i] = *(const float4*)(emb + (size_t)(m0 + row) * K + k0 + BK + q);
        rb[i] = *(const float4*)(lut + (size_t)gr * K + k0 + BK + q);
      }
    }

    v16bf ah[4], al[4], bh[2], bl[2];
#pragma unroll
    for (int i = 0; i < 4; ++i) {
      ah[i] = frag_a16(sAh, mw + i * 16, lane);
      al[i] = frag_a16(sAl, mw + i * 16, lane);
    }
#pragma unroll
    for (int j = 0; j < 2; ++j) {
      bh[j] = frag_b16(sBh, nw + j * 16, lane);
      bl[j] = frag_b16(sBl, nw + j * 16, lane);
    }
#pragma unroll
    for (int i = 0; i < 4; ++i)
#pragma unroll
      for (int j = 0; j < 2; ++j) acc[i][j] = wmma3(ah[i], al[i], bh[j], bl[j], acc[i][j]);
    __syncthreads();
  }

  // Epilogue: squared distance + min reduce into LDS, then one atomic per row.
#pragma unroll
  for (int i = 0; i < 4; ++i)
#pragma unroll
    for (int r = 0; r < 8; ++r) {
      int   rowL = mw + i * 16 + r + ((lane >> 4) << 3);
      float en   = enorm[m0 + rowL];
#pragma unroll
      for (int j = 0; j < 2; ++j) {
        int   gcol = n0 + nw + j * 16 + (lane & 15);
        float ln   = (gcol < N) ? lnorm[gcol] : __uint_as_float(INF_BITS);
        float sq   = fmaxf(en + ln - 2.0f * acc[i][j][r], 0.0f);
        atomicMin(&smin[rowL], __float_as_uint(sq));  // sq >= 0: bits order = float order
      }
    }
  __syncthreads();
  if (tid < BM) atomicMin(&minBits[m0 + tid], smin[tid]);
}

__global__ void k_final(const unsigned* __restrict__ minBits,
                        const float* __restrict__ mseAcc,
                        float* __restrict__ out, int B, float invCount) {
  int b = blockIdx.x * blockDim.x + threadIdx.x;
  if (b < B) {
    float mse = *mseAcc * invCount;
    float d   = sqrtf(__uint_as_float(minBits[b]));
    out[b]    = mse + fminf(50.0f - d, 0.0f) * 5.0f;
  }
}

// ---------------------------------------------------------------------------
extern "C" void kernel_launch(void* const* d_in, const int* in_sizes, int n_in,
                              void* d_out, int out_size, void* d_ws, size_t ws_size,
                              hipStream_t stream) {
  const float* x   = (const float*)d_in[0];
  const float* y   = (const float*)d_in[1];
  const float* W   = (const float*)d_in[2];
  const float* lut = (const float*)d_in[3];
  float* out = (float*)d_out;

  const int F = 3 * 64 * 64;            // 12288
  const int D = 512;
  const int B = in_sizes[0] / F;        // 2048
  const int N = in_sizes[3] / D;        // 100000

  // Workspace layout (~4.5 MB)
  float*    emb     = (float*)d_ws;
  float*    enorm   = emb + (size_t)B * D;
  float*    lnorm   = enorm + B;
  unsigned* minBits = (unsigned*)(lnorm + N);
  float*    mseAcc  = (float*)(minBits + B);

  k_init<<<(B + 255) / 256, 256, 0, stream>>>(minBits, mseAcc, B);
  k_mse<<<1024, 256, 0, stream>>>((const float4*)x, (const float4*)y, mseAcc,
                                  (long long)B * F / 4);
  k_rownorm<<<N, 128, 0, stream>>>(lut, lnorm, D);
  k_gemm1<<<dim3(B / BM, D / BN), 256, 0, stream>>>(x, W, emb, B, D, F);
  k_rownorm<<<B, 128, 0, stream>>>(emb, enorm, D);
  k_dist<<<dim3(B / BM, (N + BN - 1) / BN), 256, 0, stream>>>(emb, lut, enorm, lnorm,
                                                              minBits, B, N, D);
  k_final<<<(B + 255) / 256, 256, 0, stream>>>(minBits, mseAcc, out, B,
                                               1.0f / ((float)B * (float)F));
}
